// MoLE_68659347194411
// MI455X (gfx1250) — matrix operand order repaired and checked by reading, not compiled, not run on hardware
//
#include <hip/hip_runtime.h>
#include <math.h>

typedef __attribute__((ext_vector_type(16))) _Float16 v16h;
typedef __attribute__((ext_vector_type(8)))  float    v8f;
typedef __attribute__((ext_vector_type(4)))  float    f32x4;
typedef __attribute__((ext_vector_type(4)))  unsigned u32x4;

#define BATCH  2
#define DIMC   128
#define HALFC  64
#define DINNER 256
#define DSTATE 16
#define DTRANK 8
#define KDIRS  4
#define MLPHID 512
#define HH     64
#define WW     96
#define LLEN   (HH*WW)        // 6144
#define BLTOK  (BATCH*LLEN)   // 12288
#define BIGM   0x40000000

// activation codes: 0 none, 1 relu6, 2 gelu(exact), 3 softplus, 4 silu
__device__ __forceinline__ float actf(float x, int act) {
  switch (act) {
    case 1: return fminf(fmaxf(x, 0.f), 6.f);
    case 2: return 0.5f * x * (1.f + erff(x * 0.70710678118f));
    case 3: return (x > 20.f) ? x : log1pf(__expf(x));
    case 4: return x / (1.f + __expf(-x));
    default: return x;
  }
}

__device__ __forceinline__ unsigned pack2h(float a, float b) {
  union { _Float16 h[2]; unsigned u; } t;
  t.h[0] = (_Float16)a; t.h[1] = (_Float16)b;
  return t.u;
}

// ---------------------------------------------------------------------------
// Generic strided, batched WMMA GEMM:  C = act(A(MxK) * B(KxN) + bias) [+ C]
// Block tile 64(M) x 128(N), K-step 32, 256 threads = 8 waves in a 4x2 grid.
// Each wave owns a 16(M) x 64(N) strip: 4 WMMAs per K-step sharing one A
// fragment (reuse_a hint set on the first three). LDS holds f16 pairs packed
// in uint32 so each WMMA source VGPR is exactly one LDS word.
// ---------------------------------------------------------------------------
#define TM 64
#define TN 128
#define TK 32
#define AP 20    // A LDS row stride in uint32 (16 K-pairs + pad, mult of 4)
#define BP 132   // B LDS row stride in uint32 (128 n + pad, mult of 4)

__global__ __launch_bounds__(256) void wmma_gemm(
    const float* __restrict__ A, const float* __restrict__ B,
    const float* __restrict__ bias, float* __restrict__ C,
    int M, int N, int Kd,
    long long ar, long long ac, long long abs_, int abm,
    long long br, long long bc, long long bbs, int bbm,
    long long cr, long long cc, long long cbs, int cbm,
    long long biasbs, int biasbm, int bias_mode,
    int beta, int act)
{
  __shared__ unsigned As2[TM * AP];         // [m][kpair]        5120 B
  __shared__ unsigned Bs2[(TK / 2) * BP];   // [kpair][n]        8448 B
  const int tid  = threadIdx.x;
  const int lane = tid & 31;
  const int wv   = tid >> 5;
  const int wm   = wv >> 1;   // 0..3
  const int wn   = wv & 1;    // 0..1
  const int bt   = blockIdx.z;
  const int mBase = blockIdx.y * TM;
  const int nBase = blockIdx.x * TN;
  const float* Ab = A + (long long)(bt % abm) * abs_;
  const float* Bb = B + (long long)(bt % bbm) * bbs;
  float*       Cb = C + (long long)(bt % cbm) * cbs;
  const float* biasb = bias ? (bias + (long long)(bt % biasbm) * biasbs) : nullptr;

  const bool kmul = (Kd & (TK - 1)) == 0;
  const bool ainb = kmul && (mBase + TM <= M);
  const bool binb = kmul && (nBase + TN <= N);
  const int amode = !ainb ? 0 :
      ((ac == 1 && (ar & 3) == 0 && ((((unsigned long long)(const void*)Ab) & 15) == 0)) ? 1 :
       ((ar == 1 && (ac & 3) == 0 && ((((unsigned long long)(const void*)Ab) & 15) == 0)) ? 2 : 0));
  const int bmode = (binb && bc == 1 && (br & 3) == 0 &&
                     ((((unsigned long long)(const void*)Bb) & 15) == 0)) ? 1 : 0;

  v8f acc[4] = {};
  const int l16 = lane & 15;
  const int hi  = lane >> 4;

  for (int kk = 0; kk < Kd; kk += TK) {
    __syncthreads();
    // ---- stage A tile (64 x 32) as 16 K-pairs per row ----
    if (amode == 1) {           // unit K stride: 2x float4 loads along K
      const int r = tid >> 2, cg = tid & 3;
      const float* p = Ab + (long long)(mBase + r) * ar + kk + cg * 8;
      f32x4 v0 = *(const f32x4*)p;
      f32x4 v1 = *(const f32x4*)(p + 4);
      if (kk + TK < Kd) __builtin_prefetch(p + TK, 0, 3);
      u32x4 q = { pack2h(v0[0], v0[1]), pack2h(v0[2], v0[3]),
                  pack2h(v1[0], v1[1]), pack2h(v1[2], v1[3]) };
      *reinterpret_cast<u32x4*>(&As2[r * AP + cg * 4]) = q;
    } else if (amode == 2) {    // unit M stride (column-major A): float4 loads along M
      const int kp = tid & 15, m0 = (tid >> 4) * 4;
      const int k0 = kk + kp * 2;
      const float* p0 = Ab + (long long)(mBase + m0) + (long long)k0 * ac;
      f32x4 a0 = *(const f32x4*)p0;
      f32x4 a1 = *(const f32x4*)(p0 + ac);
      if (kk + TK < Kd) __builtin_prefetch(p0 + (long long)TK * ac, 0, 3);
      As2[(m0 + 0) * AP + kp] = pack2h(a0[0], a1[0]);
      As2[(m0 + 1) * AP + kp] = pack2h(a0[1], a1[1]);
      As2[(m0 + 2) * AP + kp] = pack2h(a0[2], a1[2]);
      As2[(m0 + 3) * AP + kp] = pack2h(a0[3], a1[3]);
    } else {                    // guarded generic path (branchless: clamp + select)
      #pragma unroll
      for (int i = 0; i < 4; ++i) {
        int idx = tid + 256 * i;
        int m = idx >> 4, kp = idx & 15;
        int gm = mBase + m, k0 = kk + 2 * kp;
        int gmc = gm < M ? gm : M - 1;
        int k0c = k0 < Kd ? k0 : Kd - 1;
        int k1c = (k0 + 1) < Kd ? (k0 + 1) : Kd - 1;
        float x = Ab[(long long)gmc * ar + (long long)k0c * ac];
        float y = Ab[(long long)gmc * ar + (long long)k1c * ac];
        if (!(gm < M && k0 < Kd))     x = 0.f;
        if (!(gm < M && k0 + 1 < Kd)) y = 0.f;
        As2[m * AP + kp] = pack2h(x, y);
      }
    }
    // ---- stage B tile (32 x 128) as 16 K-pairs x 128 n ----
    if (bmode == 1) {           // unit N stride: float4 loads along N, pair two K rows
      const int kp = tid >> 4, n0 = (tid & 15) * 8;
      const float* r0 = Bb + (long long)(kk + 2 * kp) * br + nBase + n0;
      const float* r1 = r0 + br;
      f32x4 x0a = *(const f32x4*)r0;
      f32x4 x0b = *(const f32x4*)(r0 + 4);
      f32x4 x1a = *(const f32x4*)r1;
      f32x4 x1b = *(const f32x4*)(r1 + 4);
      if (kk + TK < Kd) __builtin_prefetch(r0 + (long long)TK * br, 0, 3);
      u32x4 qa = { pack2h(x0a[0], x1a[0]), pack2h(x0a[1], x1a[1]),
                   pack2h(x0a[2], x1a[2]), pack2h(x0a[3], x1a[3]) };
      u32x4 qb = { pack2h(x0b[0], x1b[0]), pack2h(x0b[1], x1b[1]),
                   pack2h(x0b[2], x1b[2]), pack2h(x0b[3], x1b[3]) };
      *reinterpret_cast<u32x4*>(&Bs2[kp * BP + n0])     = qa;
      *reinterpret_cast<u32x4*>(&Bs2[kp * BP + n0 + 4]) = qb;
    } else {
      #pragma unroll
      for (int i = 0; i < 8; ++i) {
        int idx = tid + 256 * i;
        int kp = idx >> 7, n = idx & 127;
        int gn = nBase + n, k0 = kk + 2 * kp;
        int gnc = gn < N ? gn : N - 1;
        int k0c = k0 < Kd ? k0 : Kd - 1;
        int k1c = (k0 + 1) < Kd ? (k0 + 1) : Kd - 1;
        float x = Bb[(long long)k0c * br + (long long)gnc * bc];
        float y = Bb[(long long)k1c * br + (long long)gnc * bc];
        if (!(gn < N && k0 < Kd))     x = 0.f;
        if (!(gn < N && k0 + 1 < Kd)) y = 0.f;
        Bs2[kp * BP + n] = pack2h(x, y);
      }
    }
    __syncthreads();

    // ---- assemble fragments (each source VGPR = one 32-bit LDS word) ----
    union { unsigned u[8]; v16h v; } ua, ub;
    #pragma unroll
    for (int v = 0; v < 8; ++v)   // A kpair = hi*8 + v
      ua.u[v] = As2[(wm * 16 + l16) * AP + (hi << 3) + v];

    // 4 WMMAs share the A fragment; reuse_a hint on the first three.
    #pragma unroll
    for (int v = 0; v < 8; ++v)   // B kpair = hi*8 + v
      ub.u[v] = Bs2[((hi << 3) + v) * BP + wn * 64 + 0 * 16 + l16];
    acc[0] = __builtin_amdgcn_wmma_f32_16x16x32_f16(false, ua.v, false, ub.v,
                                                    (short)0, acc[0], true, false);
    #pragma unroll
    for (int v = 0; v < 8; ++v)
      ub.u[v] = Bs2[((hi << 3) + v) * BP + wn * 64 + 1 * 16 + l16];
    acc[1] = __builtin_amdgcn_wmma_f32_16x16x32_f16(false, ua.v, false, ub.v,
                                                    (short)0, acc[1], true, false);
    #pragma unroll
    for (int v = 0; v < 8; ++v)
      ub.u[v] = Bs2[((hi << 3) + v) * BP + wn * 64 + 2 * 16 + l16];
    acc[2] = __builtin_amdgcn_wmma_f32_16x16x32_f16(false, ua.v, false, ub.v,
                                                    (short)0, acc[2], true, false);
    #pragma unroll
    for (int v = 0; v < 8; ++v)
      ub.u[v] = Bs2[((hi << 3) + v) * BP + wn * 64 + 3 * 16 + l16];
    acc[3] = __builtin_amdgcn_wmma_f32_16x16x32_f16(false, ua.v, false, ub.v,
                                                    (short)0, acc[3], false, false);
  }

  #pragma unroll
  for (int j = 0; j < 4; ++j) {
    const int n_g = nBase + wn * 64 + j * 16 + l16;
    #pragma unroll
    for (int r = 0; r < 8; ++r) {
      int m_g = mBase + wm * 16 + r + (hi << 3);
      if (m_g < M && n_g < N) {
        float val = acc[j][r];
        if (bias_mode == 1)      val += biasb[n_g];
        else if (bias_mode == 2) val += biasb[m_g];
        val = actf(val, act);
        long long ci = (long long)m_g * cr + (long long)n_g * cc;
        if (beta) val += Cb[ci];
        Cb[ci] = val;
      }
    }
  }
}

// ---------------------------------------------------------------------------
// Depthwise 3x3 conv, reflect(1) or zero(0) padding, fused bias + activation.
// Layout (BATCH, C, HH, WW).
// ---------------------------------------------------------------------------
__global__ void dwconv3x3(const float* __restrict__ in, const float* __restrict__ w,
                          const float* __restrict__ bias, float* __restrict__ out,
                          int C, int reflect, int act)
{
  long long idx = (long long)blockIdx.x * 256 + threadIdx.x;
  long long total = (long long)BATCH * C * LLEN;
  if (idx >= total) return;
  int x = (int)(idx % WW);
  int y = (int)((idx / WW) % HH);
  int c = (int)((idx / LLEN) % C);
  int b = (int)(idx / ((long long)C * LLEN));
  const float* wp = w + c * 9;
  const float* ip = in + ((long long)(b * C + c)) * LLEN;
  float acc = 0.f;
  #pragma unroll
  for (int dy = -1; dy <= 1; ++dy)
    #pragma unroll
    for (int dx = -1; dx <= 1; ++dx) {
      int yy = y + dy, xx = x + dx;
      if (reflect) {
        yy = (yy < 0) ? 1 : ((yy >= HH) ? HH - 2 : yy);
        xx = (xx < 0) ? 1 : ((xx >= WW) ? WW - 2 : xx);
        acc += ip[yy * WW + xx] * wp[(dy + 1) * 3 + dx + 1];
      } else {
        if (yy >= 0 && yy < HH && xx >= 0 && xx < WW)
          acc += ip[yy * WW + xx] * wp[(dy + 1) * 3 + dx + 1];
      }
    }
  if (bias) acc += bias[c];
  out[idx] = actf(acc, act);
}

// ---------------------------------------------------------------------------
// LayerNorm over last dim C (rows = grid.x), eps 1e-5, biased variance.
// ---------------------------------------------------------------------------
__global__ __launch_bounds__(256) void layernorm_k(const float* __restrict__ in,
    float* __restrict__ out, const float* __restrict__ g, const float* __restrict__ b, int C)
{
  __shared__ float red[256];
  const int row = blockIdx.x;
  const float* ip = in + (long long)row * C;
  float s = 0.f;
  for (int c = threadIdx.x; c < C; c += 256) s += ip[c];
  red[threadIdx.x] = s; __syncthreads();
  for (int o = 128; o > 0; o >>= 1) {
    if (threadIdx.x < o) red[threadIdx.x] += red[threadIdx.x + o];
    __syncthreads();
  }
  float mu = red[0] / C;
  __syncthreads();
  float s2 = 0.f;
  for (int c = threadIdx.x; c < C; c += 256) { float d = ip[c] - mu; s2 += d * d; }
  red[threadIdx.x] = s2; __syncthreads();
  for (int o = 128; o > 0; o >>= 1) {
    if (threadIdx.x < o) red[threadIdx.x] += red[threadIdx.x + o];
    __syncthreads();
  }
  float rstd = rsqrtf(red[0] / C + 1e-5f);
  float* op = out + (long long)row * C;
  for (int c = threadIdx.x; c < C; c += 256)
    op[c] = (ip[c] - mu) * rstd * g[c] + b[c];
}

// ---------------------------------------------------------------------------
// Selective scan: one thread per (b,k,d), h[16]+A[16] in registers,
// B/C staged through LDS in 64-step chunks. Writes ys over the dt buffer.
// ---------------------------------------------------------------------------
__global__ __launch_bounds__(256) void scan_k(const float* __restrict__ xs,
    float* __restrict__ dty, const float* __restrict__ xdbl,
    const float* __restrict__ A_log, const float* __restrict__ Dp)
{
  const int bk = blockIdx.x;
  const int k  = bk & (KDIRS - 1);
  const int d  = threadIdx.x;
  const float* u  = xs   + ((long long)bk * DINNER + d) * LLEN;
  float*       y  = dty  + ((long long)bk * DINNER + d) * LLEN;
  const float* Bs = xdbl + ((long long)bk * 40 + DTRANK) * LLEN;
  const float* Cs = xdbl + ((long long)bk * 40 + DTRANK + DSTATE) * LLEN;
  float a[DSTATE], h[DSTATE];
  #pragma unroll
  for (int n = 0; n < DSTATE; ++n) {
    a[n] = -__expf(A_log[((long long)k * DINNER + d) * DSTATE + n]);
    h[n] = 0.f;
  }
  const float Dv = Dp[k * DINNER + d];
  __shared__ float sB[DSTATE * 64];
  __shared__ float sC[DSTATE * 64];
  for (int l0 = 0; l0 < LLEN; l0 += 64) {
    __syncthreads();
    for (int idx = threadIdx.x; idx < DSTATE * 64; idx += 256) {
      int n = idx >> 6, j = idx & 63;
      sB[idx] = Bs[(long long)n * LLEN + l0 + j];
      sC[idx] = Cs[(long long)n * LLEN + l0 + j];
    }
    __syncthreads();
    for (int j = 0; j < 64; ++j) {
      const int l = l0 + j;
      const float dtv = y[l];
      const float uv  = u[l];
      float acc = 0.f;
      #pragma unroll
      for (int n = 0; n < DSTATE; ++n) {
        h[n] = h[n] * __expf(dtv * a[n]) + dtv * sB[(n << 6) + j] * uv;
        acc += h[n] * sC[(n << 6) + j];
      }
      y[l] = acc + uv * Dv;
    }
  }
}

// ---------------------------------------------------------------------------
// Elementwise / transpose glue kernels
// ---------------------------------------------------------------------------
__global__ void concat_k(const float* __restrict__ rgb, const float* __restrict__ irr,
                         float* __restrict__ out)
{
  long long idx = (long long)blockIdx.x * 256 + threadIdx.x;
  if (idx >= (long long)BATCH * DIMC * LLEN) return;
  int p = (int)(idx % LLEN);
  int c = (int)((idx / LLEN) % DIMC);
  int b = (int)(idx / ((long long)DIMC * LLEN));
  out[idx] = (c < HALFC) ? rgb[((long long)(b * HALFC + c)) * LLEN + p]
                         : irr[((long long)(b * HALFC + c - HALFC)) * LLEN + p];
}

__global__ void z1_update_k(float* __restrict__ z, const float* __restrict__ r,
                            const float* __restrict__ r2)
{
  long long idx = (long long)blockIdx.x * 256 + threadIdx.x;
  if (idx >= (long long)BATCH * HALFC * LLEN) return;
  long long rem = idx % ((long long)HALFC * LLEN);
  int b = (int)(idx / ((long long)HALFC * LLEN));
  float* zp = z + (long long)b * DIMC * LLEN + rem;
  *zp = *zp * __expf(r[idx]) + r2[idx];
}

__global__ void nchw_to_nlc_k(const float* __restrict__ in, float* __restrict__ out)
{
  long long idx = (long long)blockIdx.x * 256 + threadIdx.x;
  if (idx >= (long long)BATCH * DIMC * LLEN) return;
  int c = (int)(idx % DIMC);
  long long t = idx / DIMC;
  int p = (int)(t % LLEN);
  int b = (int)(t / LLEN);
  out[idx] = in[((long long)(b * DIMC + c)) * LLEN + p];
}

__global__ void nlc_to_nchw_k(const float* __restrict__ in, float* __restrict__ out,
                              int Cc, int S, int off)
{
  long long idx = (long long)blockIdx.x * 256 + threadIdx.x;
  if (idx >= (long long)BATCH * Cc * LLEN) return;
  int p = (int)(idx % LLEN);
  int c = (int)((idx / LLEN) % Cc);
  int b = (int)(idx / ((long long)Cc * LLEN));
  out[idx] = in[((long long)b * LLEN + p) * S + off + c];
}

__global__ void nlc_add_to_nchw_k(const float* __restrict__ a, const float* __restrict__ yold,
                                  float* __restrict__ out)
{
  long long idx = (long long)blockIdx.x * 256 + threadIdx.x;
  if (idx >= (long long)BATCH * DIMC * LLEN) return;
  int p = (int)(idx % LLEN);
  int c = (int)((idx / LLEN) % DIMC);
  int b = (int)(idx / ((long long)DIMC * LLEN));
  out[idx] = yold[idx] + a[((long long)b * LLEN + p) * DIMC + c];
}

__global__ void build_xs_k(const float* __restrict__ xc, float* __restrict__ xs)
{
  long long idx = (long long)blockIdx.x * 256 + threadIdx.x;
  if (idx >= (long long)BATCH * DINNER * LLEN) return;
  int l = (int)(idx % LLEN);
  int d = (int)((idx / LLEN) % DINNER);
  int b = (int)(idx / ((long long)DINNER * LLEN));
  const float* src = xc + ((long long)(b * DINNER + d)) * LLEN;
  long long base = (((long long)b * KDIRS) * DINNER + d) * LLEN + l;
  const long long S = (long long)DINNER * LLEN;
  int lt  = (l % HH) * WW + (l / HH);          // (W,H)-major traversal
  int lr  = LLEN - 1 - l;
  int ltr = (lr % HH) * WW + (lr / HH);
  xs[base]         = src[l];
  xs[base + S]     = src[lt];
  xs[base + 2 * S] = src[lr];
  xs[base + 3 * S] = src[ltr];
}

__global__ void ysum_k(const float* __restrict__ ys, float* __restrict__ ytok)
{
  long long idx = (long long)blockIdx.x * 256 + threadIdx.x;
  if (idx >= (long long)BATCH * DINNER * LLEN) return;
  int l = (int)(idx % LLEN);
  int d = (int)((idx / LLEN) % DINNER);
  int b = (int)(idx / ((long long)DINNER * LLEN));
  const float* p0 = ys + (((long long)b * KDIRS) * DINNER + d) * LLEN;
  const long long S = (long long)DINNER * LLEN;
  int l1 = (l % WW) * HH + (l / WW);
  int l2 = LLEN - 1 - l;
  int l3 = LLEN - 1 - l1;
  float v = p0[l] + p0[S + l1] + p0[2 * S + l2] + p0[3 * S + l3];
  ytok[((long long)(b * LLEN + l)) * DINNER + d] = v;
}

__global__ void silu_mul_k(float* __restrict__ y, const float* __restrict__ xz)
{
  long long idx = (long long)blockIdx.x * 256 + threadIdx.x;
  if (idx >= (long long)BLTOK * DINNER) return;
  int c = (int)(idx % DINNER);
  long long t = idx / DINNER;
  float z = xz[t * (2 * DINNER) + DINNER + c];
  y[idx] *= z / (1.f + __expf(-z));
}

__global__ void gelu_k(float* __restrict__ x, long long total)
{
  long long idx = (long long)blockIdx.x * 256 + threadIdx.x;
  if (idx >= total) return;
  x[idx] = actf(x[idx], 2);
}

// ---------------------------------------------------------------------------
// Host side
// ---------------------------------------------------------------------------
struct IRB { const float *w1, *w2, *w3; };
struct DET { const float *sw, *sb; IRB phi, rho, eta; };

static inline dim3 ewg(long long n) { return dim3((unsigned)((n + 255) / 256)); }

static void launch_gemm(hipStream_t s, const float* A, const float* B,
    const float* bias, float* C, int M, int N, int K,
    long long ar, long long ac, long long abs_, int abm,
    long long br, long long bc, long long bbs, int bbm,
    long long cr, long long cc, long long cbs, int cbm,
    long long biasbs, int biasbm, int bias_mode, int beta, int act, int batches)
{
  dim3 grid((N + TN - 1) / TN, (M + TM - 1) / TM, batches);
  wmma_gemm<<<grid, 256, 0, s>>>(A, B, bias, C, M, N, K,
      ar, ac, abs_, abm, br, bc, bbs, bbm, cr, cc, cbs, cbm,
      biasbs, biasbm, bias_mode, beta, act);
}

static void inv_res_launch(hipStream_t s, const float* src, long long src_bs,
    const IRB& p, float* dst, long long dst_bs, int beta, float* th, float* th2)
{
  // h = relu6(w1 * src)  : (B,128,L)
  launch_gemm(s, p.w1, src, nullptr, th,
      2 * HALFC, LLEN, HALFC,
      HALFC, 1, 0, 1,
      LLEN, 1, src_bs, BIGM,
      LLEN, 1, (long long)2 * HALFC * LLEN, BIGM,
      0, 1, 0, 0, 1, BATCH);
  // depthwise 3x3 reflect + relu6
  dwconv3x3<<<ewg((long long)BATCH * 2 * HALFC * LLEN), 256, 0, s>>>(
      th, p.w2, nullptr, th2, 2 * HALFC, 1, 1);
  // dst (+)= w3 * h2 : (B,64,L)
  launch_gemm(s, p.w3, th2, nullptr, dst,
      HALFC, LLEN, 2 * HALFC,
      2 * HALFC, 1, 0, 1,
      LLEN, 1, (long long)2 * HALFC * LLEN, BIGM,
      LLEN, 1, dst_bs, BIGM,
      0, 1, 0, beta, 0, BATCH);
}

extern "C" void kernel_launch(void* const* d_in, const int* in_sizes, int n_in,
                              void* d_out, int out_size, void* d_ws, size_t ws_size,
                              hipStream_t stream)
{
  (void)in_sizes; (void)n_in; (void)out_size; (void)ws_size;
  int pi = 0;
  auto nxt = [&]() { return (const float*)d_in[pi++]; };
  const float* rgb = nxt();
  const float* irr = nxt();
  DET det[2];
  for (int l = 0; l < 2; ++l) {
    det[l].sw = nxt(); det[l].sb = nxt();
    det[l].phi.w1 = nxt(); det[l].phi.w2 = nxt(); det[l].phi.w3 = nxt();
    det[l].rho.w1 = nxt(); det[l].rho.w2 = nxt(); det[l].rho.w3 = nxt();
    det[l].eta.w1 = nxt(); det[l].eta.w2 = nxt(); det[l].eta.w3 = nxt();
  }
  const float* ln1_g = nxt();  const float* ln1_b = nxt();
  const float* in_proj_w = nxt();
  const float* conv_w = nxt(); const float* conv_b = nxt();
  const float* x_proj_w = nxt();
  const float* dt_w = nxt();   const float* dt_b = nxt();
  const float* A_log = nxt();  const float* Dp = nxt();
  const float* onorm_g = nxt(); const float* onorm_b = nxt();
  const float* out_proj_w = nxt();
  const float* ln2_g = nxt();  const float* ln2_b = nxt();
  const float* mlp_w1 = nxt(); const float* mlp_b1 = nxt();
  const float* mlp_w2 = nxt(); const float* mlp_b2 = nxt();
  const float* fc1_w = nxt();  const float* fc1_b = nxt();
  const float* dw_w = nxt();   const float* dw_b = nxt();
  const float* fc2_w = nxt();  const float* fc2_b = nxt();

  float* ws = (float*)d_ws;
  size_t off = 0;
  auto alloc = [&](size_t n) { float* p = ws + off; off += n; return p; };
  float* buf_x    = alloc((size_t)BATCH * DIMC * LLEN);
  float* buf_z    = alloc((size_t)BATCH * DIMC * LLEN);
  float* buf_h    = alloc((size_t)BATCH * 2 * HALFC * LLEN);
  float* buf_h2   = alloc((size_t)BATCH * 2 * HALFC * LLEN);
  float* buf_r    = alloc((size_t)BATCH * HALFC * LLEN);
  float* buf_r2   = alloc((size_t)BATCH * HALFC * LLEN);
  float* buf_xz   = alloc((size_t)BLTOK * 2 * DINNER);
  float* buf_xm   = alloc((size_t)BATCH * DINNER * LLEN);
  float* buf_xc   = alloc((size_t)BATCH * DINNER * LLEN);
  float* buf_xs   = alloc((size_t)BATCH * KDIRS * DINNER * LLEN);
  float* buf_xdbl = alloc((size_t)BATCH * KDIRS * 40 * LLEN);
  float* buf_dt   = alloc((size_t)BATCH * KDIRS * DINNER * LLEN);
  float* buf_ytok = alloc((size_t)BLTOK * DINNER);
  // temporal aliases (lifetimes don't overlap)
  float* buf_a   = buf_z;    // token-major (BL,128) after detail path
  float* buf_t   = buf_h;    // layernorm scratch
  float* buf_mh  = buf_xs;   // MoLE hidden NCHW (B,512,L)
  float* buf_mh2 = buf_dt;   // MoLE hidden after dwconv
  float* outp = (float*)d_out;

  // ---- detail path (NCHW) -------------------------------------------------
  concat_k<<<ewg((long long)BATCH * DIMC * LLEN), 256, 0, stream>>>(rgb, irr, buf_x);
  const float* zin = buf_x;
  float* zping[2] = { buf_z, buf_x };
  for (int l = 0; l < 2; ++l) {
    float* zout = zping[l];
    // channel shuffle: 1x1 conv 128->128 + bias
    launch_gemm(stream, det[l].sw, zin, det[l].sb, zout,
        DIMC, LLEN, DIMC,
        DIMC, 1, 0, 1,
        LLEN, 1, (long long)DIMC * LLEN, BIGM,
        LLEN, 1, (long long)DIMC * LLEN, BIGM,
        0, 1, 2, 0, 0, BATCH);
    // z2 += inv_res(z1, phi)
    inv_res_launch(stream, zout, (long long)DIMC * LLEN, det[l].phi,
        zout + (long long)HALFC * LLEN, (long long)DIMC * LLEN, 1, buf_h, buf_h2);
    // r = inv_res(z2, rho); r2 = inv_res(z2, eta)
    inv_res_launch(stream, zout + (long long)HALFC * LLEN, (long long)DIMC * LLEN,
        det[l].rho, buf_r, (long long)HALFC * LLEN, 0, buf_h, buf_h2);
    inv_res_launch(stream, zout + (long long)HALFC * LLEN, (long long)DIMC * LLEN,
        det[l].eta, buf_r2, (long long)HALFC * LLEN, 0, buf_h, buf_h2);
    // z1 = z1*exp(r) + r2
    z1_update_k<<<ewg((long long)BATCH * HALFC * LLEN), 256, 0, stream>>>(zout, buf_r, buf_r2);
    zin = zout;
  }
  // y lives in buf_x (NCHW)

  // ---- STM block ----------------------------------------------------------
  nchw_to_nlc_k<<<ewg((long long)BATCH * DIMC * LLEN), 256, 0, stream>>>(buf_x, buf_a);
  layernorm_k<<<BLTOK, 256, 0, stream>>>(buf_a, buf_t, ln1_g, ln1_b, DIMC);
  // xz = ln1(x) @ in_proj_w   (BL x 512)
  launch_gemm(stream, buf_t, in_proj_w, nullptr, buf_xz,
      BLTOK, 2 * DINNER, DIMC,
      DIMC, 1, 0, 1,
      2 * DINNER, 1, 0, 1,
      2 * DINNER, 1, 0, 1,
      0, 1, 0, 0, 0, 1);
  // xm -> NCHW, dw-conv + bias + silu
  nlc_to_nchw_k<<<ewg((long long)BATCH * DINNER * LLEN), 256, 0, stream>>>(
      buf_xz, buf_xm, DINNER, 2 * DINNER, 0);
  dwconv3x3<<<ewg((long long)BATCH * DINNER * LLEN), 256, 0, stream>>>(
      buf_xm, conv_w, conv_b, buf_xc, DINNER, 0, 4);
  // xs: 4 scan directions
  build_xs_k<<<ewg((long long)BATCH * DINNER * LLEN), 256, 0, stream>>>(buf_xc, buf_xs);
  // x_dbl[bk] = x_proj_w[k] (40x256) @ xs[bk] (256xL)
  launch_gemm(stream, x_proj_w, buf_xs, nullptr, buf_xdbl,
      40, LLEN, DINNER,
      DINNER, 1, (long long)40 * DINNER, KDIRS,
      LLEN, 1, (long long)DINNER * LLEN, BIGM,
      LLEN, 1, (long long)40 * LLEN, BIGM,
      0, 1, 0, 0, 0, BATCH * KDIRS);
  // dt[bk] = softplus(dt_w[k] (256x8) @ dts[bk] (8xL) + dt_b[k])
  launch_gemm(stream, dt_w, buf_xdbl, dt_b, buf_dt,
      DINNER, LLEN, DTRANK,
      DTRANK, 1, (long long)DINNER * DTRANK, KDIRS,
      LLEN, 1, (long long)40 * LLEN, BIGM,
      LLEN, 1, (long long)DINNER * LLEN, BIGM,
      DINNER, KDIRS, 2, 0, 3, BATCH * KDIRS);
  // selective scan (writes ys over buf_dt, includes +u*D)
  scan_k<<<BATCH * KDIRS, 256, 0, stream>>>(buf_xs, buf_dt, buf_xdbl, A_log, Dp);
  // combine 4 directions -> token-major (BL, 256)
  ysum_k<<<ewg((long long)BATCH * DINNER * LLEN), 256, 0, stream>>>(buf_dt, buf_ytok);
  layernorm_k<<<BLTOK, 256, 0, stream>>>(buf_ytok, buf_ytok, onorm_g, onorm_b, DINNER);
  silu_mul_k<<<ewg((long long)BLTOK * DINNER), 256, 0, stream>>>(buf_ytok, buf_xz);
  // x += y @ out_proj_w
  launch_gemm(stream, buf_ytok, out_proj_w, nullptr, buf_a,
      BLTOK, DIMC, DINNER,
      DINNER, 1, 0, 1,
      DIMC, 1, 0, 1,
      DIMC, 1, 0, 1,
      0, 1, 0, 1, 0, 1);
  // MLP: x += gelu(ln2(x) @ w1 + b1) @ w2 + b2
  layernorm_k<<<BLTOK, 256, 0, stream>>>(buf_a, buf_t, ln2_g, ln2_b, DIMC);
  launch_gemm(stream, buf_t, mlp_w1, mlp_b1, buf_xz,
      BLTOK, MLPHID, DIMC,
      DIMC, 1, 0, 1,
      MLPHID, 1, 0, 1,
      MLPHID, 1, 0, 1,
      0, 1, 1, 0, 2, 1);
  launch_gemm(stream, buf_xz, mlp_w2, mlp_b2, buf_a,
      BLTOK, DIMC, MLPHID,
      MLPHID, 1, 0, 1,
      DIMC, 1, 0, 1,
      DIMC, 1, 0, 1,
      0, 1, 1, 1, 0, 1);
  // y = transpose(a) + y_old  -> written straight into d_out (NCHW)
  nlc_add_to_nchw_k<<<ewg((long long)BATCH * DIMC * LLEN), 256, 0, stream>>>(buf_a, buf_x, outp);

  // ---- MoLE conv-MLP ------------------------------------------------------
  // h = y(NCHW, read strided) @ fc1_w + fc1_b -> token-major (BL,512)
  launch_gemm(stream, outp, fc1_w, fc1_b, buf_xz,
      LLEN, MLPHID, DIMC,
      1, LLEN, (long long)DIMC * LLEN, BIGM,
      MLPHID, 1, 0, 1,
      MLPHID, 1, (long long)LLEN * MLPHID, BIGM,
      0, 1, 1, 0, 0, BATCH);
  nlc_to_nchw_k<<<ewg((long long)BATCH * MLPHID * LLEN), 256, 0, stream>>>(
      buf_xz, buf_mh, MLPHID, MLPHID, 0);
  dwconv3x3<<<ewg((long long)BATCH * MLPHID * LLEN), 256, 0, stream>>>(
      buf_mh, dw_w, dw_b, buf_mh2, MLPHID, 0, 0);
  gelu_k<<<ewg((long long)BATCH * MLPHID * LLEN), 256, 0, stream>>>(
      buf_mh2, (long long)BATCH * MLPHID * LLEN);
  // out(NCHW, strided) += h(NCHW, read strided) @ fc2_w + fc2_b
  launch_gemm(stream, buf_mh2, fc2_w, fc2_b, outp,
      LLEN, DIMC, MLPHID,
      1, LLEN, (long long)MLPHID * LLEN, BIGM,
      DIMC, 1, 0, 1,
      1, LLEN, (long long)DIMC * LLEN, BIGM,
      0, 1, 1, 1, 0, BATCH);
}
